// SAGE_REDUN_SAMPLE_66915590472352
// MI455X (gfx1250) — matrix operand order repaired and checked by reading, not compiled, not run on hardware
//
#include <hip/hip_runtime.h>
#include <hip/hip_bf16.h>
#include <math.h>

typedef __attribute__((ext_vector_type(16))) _Float16 v16h;
typedef __attribute__((ext_vector_type(8)))  _Float16 v8h;
typedef __attribute__((ext_vector_type(8)))  float    v8f;

#define F_DIM 128   // feature width F
#define KAGG  16    // aggregation fan-in K
#define NBATCH 256  // B

// ---------------------------------------------------------------------------
// Weight f32 -> f16 conversion (runs once per launch; weights stay L2-resident)
// ---------------------------------------------------------------------------
__global__ void cvt_f32_f16_kernel(const float* __restrict__ s,
                                   _Float16* __restrict__ d, int n) {
  int i = blockIdx.x * blockDim.x + threadIdx.x;
  if (i < n) d[i] = (_Float16)s[i];
}

// ---------------------------------------------------------------------------
// Fused SAGE layer:
//   out[r, :] = relu( [ mean_k aggSrc[16r+k, :] | xSide[r, :] ] @ W(2F x F) + bias )
// Output rows are linear in r (r = b*NPB + n), and the aggregation window of
// row r is exactly source rows [16r, 16r+16) -- no div/mod needed anywhere.
//
// Block = 256 threads = 8 wave32s. Each block owns 16 output rows per tile;
// wave w computes output columns [16w, 16w+16). B panel (K=256 x 16 cols, f16)
// lives in registers, A panel (16 x 256 f16) is staged in LDS per tile.
// ---------------------------------------------------------------------------
template <typename AggT>
__global__ __launch_bounds__(256) void sage_layer_kernel(
    const AggT*     __restrict__ aggSrc,  // [rows*16, 128]
    const float*    __restrict__ xSide,   // [rows, 128]
    const _Float16* __restrict__ Wf16,    // [256, 128] row-major (K x N)
    const float*    __restrict__ bias,    // [128]
    _Float16*       __restrict__ out,     // [rows, 128]
    int tilesPerWG, int totalTiles) {
  constexpr int AS = 264;  // padded LDS row stride (halves): 528B rows avoid bank clash
  __shared__ __align__(16) _Float16 A_lds[16 * AS];

  const int tid  = threadIdx.x;
  const int lane = tid & 31;
  const int wv   = tid >> 5;      // wave id 0..7 -> N-tile
  const int hi   = lane >> 4;     // half-wave select
  const int lo16 = lane & 15;
  const int col  = wv * 16 + lo16;

  // --- B fragments (weights) in registers: 8 fragments cover K = 0..255 ---
  // 16-bit B 32x16 layout: lane half hi holds K=8hi..8hi+7 then K=16+8hi..16+8hi+7.
  v16h bfrag[8];
  for (int kk = 0; kk < 8; ++kk) {
    union { v16h v; _Float16 e[16]; } u;
    const int kb = kk * 32 + hi * 8;
#pragma unroll
    for (int j = 0; j < 16; ++j) u.e[j] = Wf16[(size_t)(kb + (j < 8 ? j : j + 8)) * F_DIM + col];
    bfrag[kk] = u.v;
  }
  const float bv = bias[col];

  for (int t = 0; t < tilesPerWG; ++t) {
    const int tile = blockIdx.x * tilesPerWG + t;
    if (tile >= totalTiles) break;
    const int r0 = tile * 16;

    // Hint-prefetch next tile's aggregation source (global_prefetch_b8).
    if (t + 1 < tilesPerWG && tile + 1 < totalTiles)
      __builtin_prefetch(aggSrc + ((size_t)(r0 + 16) * KAGG + tid) * F_DIM, 0, 0);

    // --- Stage A panel: [16 rows][K=0..127 agg | K=128..255 side] as f16 ---
    for (int p = tid; p < 16 * F_DIM; p += 256) {
      const int m = p >> 7;          // local row 0..15
      const int f = p & (F_DIM - 1); // feature
      const size_t r = (size_t)(r0 + m);
      const AggT* src = aggSrc + r * KAGG * F_DIM + f;
      float s = 0.f;
#pragma unroll
      for (int k = 0; k < KAGG; ++k) s += (float)src[(size_t)k * F_DIM];
      A_lds[m * AS + f]         = (_Float16)(s * (1.0f / KAGG));
      A_lds[m * AS + F_DIM + f] = (_Float16)xSide[r * F_DIM + f];
    }
    __syncthreads();

    // --- accumulate: C initialized with bias (bias depends on col only) ---
    v8f c;
#pragma unroll
    for (int v = 0; v < 8; ++v) c[v] = bv;

    // 16-bit A 16x32 layout: lane(lo16,hi) row=lo16; halves 0..7 = K kb..kb+7,
    // halves 8..15 = K kb+16..kb+23, with kb = 32*kk + 8*hi. Two ds_load_b128.
#pragma unroll
    for (int kk = 0; kk < 8; ++kk) {
      const int kb = kk * 32 + hi * 8;
      union { v16h v; v8h h[2]; } ua;
      ua.h[0] = *(const v8h*)&A_lds[lo16 * AS + kb];
      ua.h[1] = *(const v8h*)&A_lds[lo16 * AS + kb + 16];
      c = __builtin_amdgcn_wmma_f32_16x16x32_f16(
              /*neg_a=*/false, ua.v, /*neg_b=*/false, bfrag[kk],
              /*c_mod=*/(short)0, c, /*reuse_a=*/false, /*reuse_b=*/false);
    }

    // --- ReLU + f16 store. C layout: VGPR v, lane L -> row v+8*hi, col L%16 ---
#pragma unroll
    for (int v = 0; v < 8; ++v) {
      const size_t r = (size_t)(r0 + v + 8 * hi);
      out[r * F_DIM + col] = (_Float16)fmaxf(c[v], 0.0f);
    }
    __syncthreads();
  }
}

// ---------------------------------------------------------------------------
// Head: z = h0 @ lin1_W + b ; BatchNorm(batch stats) ; lin2 ; sigmoid
// One block of 256 threads; thread b owns batch row b. Tiny (0.5 MFLOP).
// ---------------------------------------------------------------------------
__global__ __launch_bounds__(256) void head_kernel(
    const _Float16* __restrict__ h0,     // [256,128] f16
    const float* __restrict__ lin1W,     // [128,8]
    const float* __restrict__ lin1b,     // [8]
    const float* __restrict__ gamma,     // [8]
    const float* __restrict__ beta,      // [8]
    const float* __restrict__ lin2W,     // [8,2]
    const float* __restrict__ lin2b,     // [2]
    float* __restrict__ outp) {          // [256,2]
  __shared__ float zb[NBATCH * 8];
  __shared__ float mu_s[8], rs_s[8];
  const int b = threadIdx.x;

  float z[8];
#pragma unroll
  for (int j = 0; j < 8; ++j) z[j] = lin1b[j];
  for (int f = 0; f < F_DIM; ++f) {
    const float xv = (float)h0[b * F_DIM + f];
#pragma unroll
    for (int j = 0; j < 8; ++j) z[j] += xv * lin1W[f * 8 + j];
  }
#pragma unroll
  for (int j = 0; j < 8; ++j) zb[b * 8 + j] = z[j];
  __syncthreads();

  if (threadIdx.x < 8) {
    const int j = threadIdx.x;
    float mu = 0.f;
    for (int i = 0; i < NBATCH; ++i) mu += zb[i * 8 + j];
    mu *= (1.0f / NBATCH);
    float var = 0.f;
    for (int i = 0; i < NBATCH; ++i) { float d = zb[i * 8 + j] - mu; var += d * d; }
    var *= (1.0f / NBATCH);
    mu_s[j] = mu;
    rs_s[j] = rsqrtf(var + 1e-5f);
  }
  __syncthreads();

  float zn[8];
#pragma unroll
  for (int j = 0; j < 8; ++j)
    zn[j] = (z[j] - mu_s[j]) * rs_s[j] * gamma[j] + beta[j];
#pragma unroll
  for (int c2 = 0; c2 < 2; ++c2) {
    float acc = lin2b[c2];
#pragma unroll
    for (int j = 0; j < 8; ++j) acc += zn[j] * lin2W[j * 2 + c2];
    outp[b * 2 + c2] = 1.0f / (1.0f + expf(-acc));
  }
}

// ---------------------------------------------------------------------------
extern "C" void kernel_launch(void* const* d_in, const int* in_sizes, int n_in,
                              void* d_out, int out_size, void* d_ws, size_t ws_size,
                              hipStream_t stream) {
  (void)in_sizes; (void)n_in; (void)out_size; (void)ws_size;

  const float* x0    = (const float*)d_in[0];   // [256,128]
  const float* x1    = (const float*)d_in[1];   // [256,16,128]
  const float* x2    = (const float*)d_in[2];   // [256,256,128]
  const float* x3    = (const float*)d_in[3];   // [256,4096,128]
  const float* W0    = (const float*)d_in[4];
  const float* b0    = (const float*)d_in[5];
  const float* W1    = (const float*)d_in[6];
  const float* b1    = (const float*)d_in[7];
  const float* W2    = (const float*)d_in[8];
  const float* b2    = (const float*)d_in[9];
  const float* lin1W = (const float*)d_in[10];
  const float* lin1b = (const float*)d_in[11];
  const float* gamma = (const float*)d_in[12];
  const float* beta  = (const float*)d_in[13];
  const float* lin2W = (const float*)d_in[14];
  const float* lin2b = (const float*)d_in[15];

  // Workspace layout (bytes)
  char* ws = (char*)d_ws;
  _Float16* W2f = (_Float16*)(ws + 0);                        // 64 KB
  _Float16* W1f = (_Float16*)(ws + (64u << 10));              // 64 KB
  _Float16* W0f = (_Float16*)(ws + (128u << 10));             // 64 KB
  _Float16* h2  = (_Float16*)(ws + (192u << 10));             // 16 MB  [256*256,128]
  _Float16* h1  = (_Float16*)(ws + (192u << 10) + (16u << 20));   // 1 MB [256*16,128]
  _Float16* h0  = (_Float16*)(ws + (192u << 10) + (17u << 20));   // 64 KB [256,128]

  const int wN = 2 * F_DIM * F_DIM;  // 32768 per weight matrix
  cvt_f32_f16_kernel<<<(wN + 255) / 256, 256, 0, stream>>>(W2, W2f, wN);
  cvt_f32_f16_kernel<<<(wN + 255) / 256, 256, 0, stream>>>(W1, W1f, wN);
  cvt_f32_f16_kernel<<<(wN + 255) / 256, 256, 0, stream>>>(W0, W0f, wN);

  // Layer 3: rows = 256*256 = 65536 -> 4096 tiles, 8 tiles/block -> 512 blocks
  sage_layer_kernel<float><<<512, 256, 0, stream>>>(
      x3, x2, W2f, b2, h2, /*tilesPerWG=*/8, /*totalTiles=*/4096);

  // Layer 2: rows = 256*16 = 4096 -> 256 tiles, 1 tile/block
  sage_layer_kernel<_Float16><<<256, 256, 0, stream>>>(
      h2, x1, W1f, b1, h1, 1, 256);

  // Layer 1: rows = 256 -> 16 tiles, 1 tile/block
  sage_layer_kernel<_Float16><<<16, 256, 0, stream>>>(
      h1, x0, W0f, b0, h0, 1, 16);

  head_kernel<<<1, 256, 0, stream>>>(h0, lin1W, lin1b, gamma, beta,
                                     lin2W, lin2b, (float*)d_out);
}
